// CrossAttention_Casual_88837103550790
// MI455X (gfx1250) — matrix operand order repaired and checked by reading, not compiled, not run on hardware
//
#include <hip/hip_runtime.h>
#include <hip/hip_bf16.h>
#include <stdint.h>

// ---------------- problem constants ----------------
#define BATCH   4
#define CDIM    384
#define HEADS   6
#define DHEAD   64
#define NPIX    1024                 // 32*32 tokens
#define INNER   384                  // HEADS*DHEAD
#define BH      (BATCH * HEADS)      // 24
#define NN      (NPIX * NPIX)        // 1048576
#define KTOP    734003u              // int(0.7 * 1024 * 1024)
#define SCALE_F 0.125f               // 64^-0.5

typedef __bf16 bf16_t;
typedef __attribute__((ext_vector_type(16))) __bf16 v16bf;
typedef __attribute__((ext_vector_type(8)))  __bf16 v8bf;
typedef __attribute__((ext_vector_type(8)))  float  v8f;

// float -> bf16 (round to nearest even) via bit ops
__device__ __forceinline__ bf16_t f2bf(float x) {
  unsigned u = __float_as_uint(x);
  unsigned short h = (unsigned short)((u + 0x7FFFu + ((u >> 16) & 1u)) >> 16);
  return __builtin_bit_cast(bf16_t, h);
}

// ---------------------------------------------------------------------------
// LDS panels live in a per-lane fragment layout (contiguous k = 16h + s): each lane's
// 16 bf16 fragment elements are 32 contiguous bytes (2x ds_load_b128).
//   A 16x32 tile: lane = row + 16*(k>>4);     e-group g covers k=16*(lane>>4)+8g+0..7
//   B 32x16 tile: lane = col + 16*(k>>4);     e-group g covers k=16*(lane>>4)+8g+0..7
// A panel = T row-tiles x 32k, B panel = T col-tiles x 32k; tile = 512 bf16 (1KB).
// ---------------------------------------------------------------------------

template <int T>
__device__ __forceinline__ void stage_A(bf16_t* __restrict__ As,
                                        const float* __restrict__ src, int ldA) {
#pragma unroll
  for (int j = threadIdx.x; j < T * 64; j += 256) {
    const int tile = j >> 6;
    const int lane = (j >> 1) & 31;
    const int g = j & 1;
    const int row = tile * 16 + (lane & 15);
    const int k0 = ((lane >> 4) << 4) + g * 8;
    const float* s = src + (size_t)row * ldA + k0;
    v8bf v;
#pragma unroll
    for (int e = 0; e < 8; ++e) v[e] = f2bf(s[e]);
    *(v8bf*)(As + tile * 512 + lane * 16 + g * 8) = v;   // 16B ds_store
  }
}

template <int T>
__device__ __forceinline__ void stage_B_kcontig(bf16_t* __restrict__ Bs,
                                                const float* __restrict__ src,
                                                int ldB) {
#pragma unroll
  for (int j = threadIdx.x; j < T * 64; j += 256) {
    const int tc = j >> 6;
    const int lane = (j >> 1) & 31;
    const int g = j & 1;
    const int col = tc * 16 + (lane & 15);
    const int k0 = ((lane >> 4) << 4) + g * 8;
    const float* s = src + (size_t)col * ldB + k0;
    v8bf v;
#pragma unroll
    for (int e = 0; e < 8; ++e) v[e] = f2bf(s[e]);
    *(v8bf*)(Bs + tc * 512 + lane * 16 + g * 8) = v;
  }
}

// B panel from row-major [k][n] source (transpose scatter; reads coalesced).
template <int T>
__device__ __forceinline__ void stage_B_transpose(bf16_t* __restrict__ Bs,
                                                  const float* __restrict__ src,
                                                  int ldN) {
  for (int i = threadIdx.x; i < 32 * T * 16; i += 256) {
    const int k = i / (T * 16), n = i % (T * 16);
    const int tc = n >> 4;
    const int lane = (n & 15) + ((k >> 4) << 4);
    Bs[tc * 512 + lane * 16 + (k & 15)] = f2bf(src[(size_t)k * ldN + n]);
  }
}

__device__ __forceinline__ v8f wmma_frag(const bf16_t* __restrict__ Af,
                                         const bf16_t* __restrict__ Bf, v8f acc) {
  const int lane = threadIdx.x & 31;
  v16bf a = *(const v16bf*)(Af + lane * 16);
  v16bf b = *(const v16bf*)(Bf + lane * 16);
  return __builtin_amdgcn_wmma_f32_16x16x32_bf16(false, a, false, b,
                                                 (short)0, acc, false, false);
}

__device__ __forceinline__ void store_tile_rm(float* __restrict__ C, int ldc, v8f acc) {
  const int lane = threadIdx.x & 31;
  const int col  = lane & 15;
  const int r0   = (lane >> 4) * 8;
#pragma unroll
  for (int i = 0; i < 8; ++i) C[(r0 + i) * ldc + col] = acc[i];
}

// Wave tile mapping for 128x128 blocks (8x8 tiles, 8 waves, 2x4 tiles/wave):
//   row tiles: 2*(w&3), 2*(w&3)+1 ; col tiles: 4*(w>>2) .. +3

// ===========================================================================
// 1) Projections:  Out[o,n] = sum_c W[o,c] * X[b][c,n]   (M=384,K=384,N=1024)
//    grid (8, 3, 12); 128x128 blocks. Q/K stored [b,h,token,dd]; V as [b,i,m].
// ===========================================================================
__global__ __launch_bounds__(256)
void proj_gemm_kernel(const float* __restrict__ Wq, const float* __restrict__ Wk,
                      const float* __restrict__ Wv, const float* __restrict__ query,
                      const float* __restrict__ context,
                      float* __restrict__ Qf, float* __restrict__ Kf,
                      float* __restrict__ Vf) {
  const int gz = blockIdx.z;
  const int which = gz % 3, b = gz / 3;
  const float* A = (which == 0) ? Wq : (which == 1) ? Wk : Wv;
  const float* X = ((which == 0) ? query : context) + (size_t)b * CDIM * NPIX;

  const int m0 = blockIdx.y * 128;
  const int n0 = blockIdx.x * 128;
  __shared__ __align__(32) bf16_t As[2][8 * 512];
  __shared__ __align__(32) bf16_t Bs[2][8 * 512];
  const int wave = threadIdx.x >> 5;
  const int wr0 = (wave & 3) * 2, wc0 = (wave >> 2) * 4;
  v8f acc[2][4] = {};

  for (int kk = 0; kk < CDIM; kk += 64) {
    stage_A<8>(As[0], A + (size_t)m0 * CDIM + kk, CDIM);
    stage_A<8>(As[1], A + (size_t)m0 * CDIM + kk + 32, CDIM);
    stage_B_transpose<8>(Bs[0], X + (size_t)kk * NPIX + n0, NPIX);
    stage_B_transpose<8>(Bs[1], X + (size_t)(kk + 32) * NPIX + n0, NPIX);
    __syncthreads();
#pragma unroll
    for (int kp = 0; kp < 2; ++kp)
#pragma unroll
      for (int r = 0; r < 2; ++r)
#pragma unroll
        for (int c = 0; c < 4; ++c)
          acc[r][c] = wmma_frag(As[kp] + (wr0 + r) * 512,
                                Bs[kp] + (wc0 + c) * 512, acc[r][c]);
    __syncthreads();
  }

  const int lane = threadIdx.x & 31, col = lane & 15, r0 = (lane >> 4) * 8;
  if (which == 2) {
    float* Out = Vf + (size_t)b * INNER * NPIX;
#pragma unroll
    for (int r = 0; r < 2; ++r)
#pragma unroll
      for (int c = 0; c < 4; ++c)
        store_tile_rm(Out + (size_t)(m0 + (wr0 + r) * 16) * NPIX +
                          n0 + (wc0 + c) * 16, NPIX, acc[r][c]);
  } else {
    float* Outbase = (which == 0) ? Qf : Kf;
#pragma unroll
    for (int r = 0; r < 2; ++r) {
      const int mg = m0 + (wr0 + r) * 16;       // 16-row tile stays in one head
      const int h = mg >> 6, dd0 = (mg & 63) + r0;
      float* Out = Outbase + (size_t)(b * HEADS + h) * NPIX * DHEAD;
#pragma unroll
      for (int c = 0; c < 4; ++c) {
        const int nb = n0 + (wc0 + c) * 16 + col;
#pragma unroll
        for (int i = 0; i < 8; ++i)
          Out[(size_t)nb * DHEAD + dd0 + i] = acc[r][c][i];
      }
    }
  }
}

// ===========================================================================
// 2) Scores: S[n,m] = SCALE * sum_d Q[n,d]*K[m,d]; K-dim=64 -> ONE barrier window
//    grid (8, 8, 24); 128x128 blocks.
// ===========================================================================
__global__ __launch_bounds__(256)
void scores_kernel(const float* __restrict__ Qf, const float* __restrict__ Kf,
                   float* __restrict__ attn) {
  const int bh = blockIdx.z;
  const float* Q = Qf + (size_t)bh * NPIX * DHEAD;   // [n][dd]
  const float* K = Kf + (size_t)bh * NPIX * DHEAD;   // [m][dd]
  float* S = attn + (size_t)bh * NN;
  const int n0 = blockIdx.y * 128, m0 = blockIdx.x * 128;
  __shared__ __align__(32) bf16_t As[2][8 * 512];
  __shared__ __align__(32) bf16_t Bs[2][8 * 512];
  const int wave = threadIdx.x >> 5;
  const int wr0 = (wave & 3) * 2, wc0 = (wave >> 2) * 4;
  v8f acc[2][4] = {};

  stage_A<8>(As[0], Q + (size_t)n0 * DHEAD, DHEAD);
  stage_A<8>(As[1], Q + (size_t)n0 * DHEAD + 32, DHEAD);
  stage_B_kcontig<8>(Bs[0], K + (size_t)m0 * DHEAD, DHEAD);
  stage_B_kcontig<8>(Bs[1], K + (size_t)m0 * DHEAD + 32, DHEAD);
  __syncthreads();
#pragma unroll
  for (int kp = 0; kp < 2; ++kp)
#pragma unroll
    for (int r = 0; r < 2; ++r)
#pragma unroll
      for (int c = 0; c < 4; ++c)
        acc[r][c] = wmma_frag(As[kp] + (wr0 + r) * 512,
                              Bs[kp] + (wc0 + c) * 512, acc[r][c]);

#pragma unroll
  for (int r = 0; r < 2; ++r)
#pragma unroll
    for (int c = 0; c < 4; ++c) {
#pragma unroll
      for (int i = 0; i < 8; ++i) acc[r][c][i] *= SCALE_F;
      store_tile_rm(S + (size_t)(n0 + (wr0 + r) * 16) * NPIX +
                        m0 + (wc0 + c) * 16, NPIX, acc[r][c]);
    }
}

// ===========================================================================
// 3) Row softmax in place. One block (256 thr) per row of 1024.
// ===========================================================================
__global__ __launch_bounds__(256)
void softmax_kernel(float* __restrict__ attn) {
  float* p = attn + (size_t)blockIdx.x * NPIX;
  __shared__ float red[256];
  const int tid = threadIdx.x;
  float v[4], mx = -INFINITY;
#pragma unroll
  for (int j = 0; j < 4; ++j) { v[j] = p[tid + j * 256]; mx = fmaxf(mx, v[j]); }
  red[tid] = mx; __syncthreads();
  for (int s = 128; s > 0; s >>= 1) {
    if (tid < s) red[tid] = fmaxf(red[tid], red[tid + s]);
    __syncthreads();
  }
  mx = red[0]; __syncthreads();
  float sum = 0.f;
#pragma unroll
  for (int j = 0; j < 4; ++j) { v[j] = __expf(v[j] - mx); sum += v[j]; }
  red[tid] = sum; __syncthreads();
  for (int s = 128; s > 0; s >>= 1) {
    if (tid < s) red[tid] += red[tid + s];
    __syncthreads();
  }
  const float inv = 1.0f / red[0];
#pragma unroll
  for (int j = 0; j < 4; ++j) p[tid + j * 256] = v[j] * inv;
}

// ===========================================================================
// 4) Global top-k per (b,h): MSB-first radix select on float bit patterns.
// ===========================================================================
__global__ void topk_init_kernel(unsigned* hist, unsigned* pref, unsigned* want,
                                 unsigned* tie) {
  const int i = blockIdx.x * 256 + threadIdx.x;
  if (i < BH * 256) hist[i] = 0u;
  if (i < BH) { pref[i] = 0u; want[i] = KTOP; tie[i] = 0u; }
}

__global__ __launch_bounds__(256)
void topk_hist_kernel(const float* __restrict__ attn, unsigned* __restrict__ hist,
                      const unsigned* __restrict__ pref, unsigned maskTop, int shift) {
  const int bh = blockIdx.y;
  const size_t base = (size_t)bh * NN + (size_t)blockIdx.x * (NN / 64);
  __shared__ unsigned lh[256];
  lh[threadIdx.x] = 0u; __syncthreads();
  const unsigned p = pref[bh];
  for (int i = threadIdx.x; i < NN / 64; i += 256) {
    const unsigned bits = __float_as_uint(attn[base + i]);
    if ((bits & maskTop) == p) atomicAdd(&lh[(bits >> shift) & 0xFFu], 1u);
  }
  __syncthreads();
  if (lh[threadIdx.x]) atomicAdd(&hist[bh * 256 + threadIdx.x], lh[threadIdx.x]);
}

__global__ __launch_bounds__(256)
void topk_scan_kernel(unsigned* __restrict__ hist, unsigned* __restrict__ pref,
                      unsigned* __restrict__ want, int shift) {
  const int bh = blockIdx.x;
  if (threadIdx.x == 0) {
    unsigned w = want[bh], acc = 0u; int bsel = 0;
    for (int bin = 255; bin >= 0; --bin) {
      const unsigned c = hist[bh * 256 + bin];
      if (acc + c >= w) { bsel = bin; w -= acc; break; }
      acc += c;
    }
    pref[bh] |= ((unsigned)bsel) << shift;
    want[bh] = w;
  }
  __syncthreads();
  hist[bh * 256 + threadIdx.x] = 0u;
}

__global__ __launch_bounds__(256)
void topk_mask_kernel(const float* __restrict__ attn, const unsigned* __restrict__ pref,
                      const unsigned* __restrict__ want, unsigned* __restrict__ tie,
                      float* __restrict__ causal, float* __restrict__ noncausal,
                      float* __restrict__ bmask) {
  const int bh = blockIdx.y;
  const size_t base = (size_t)bh * NN + (size_t)blockIdx.x * (NN / 64);
  const unsigned T = pref[bh], q = want[bh];
  for (int i = threadIdx.x; i < NN / 64; i += 256) {
    const size_t g = base + i;
    const float v = attn[g];
    const unsigned bits = __float_as_uint(v);
    float m;
    if (bits > T)       m = 1.0f;
    else if (bits == T) m = (atomicAdd(&tie[bh], 1u) < q) ? 1.0f : 0.0f;
    else                m = 0.0f;
    bmask[g]     = m;
    causal[g]    = (m != 0.0f) ? v : 0.0f;
    noncausal[g] = (m != 0.0f) ? 0.0f : v;
  }
}

// ===========================================================================
// 5) attn_map[b,m] = mean over (h,n) of attn[b,h,n,m].   grid (4, 4) x 256
// ===========================================================================
__global__ __launch_bounds__(256)
void attnmap_kernel(const float* __restrict__ attn, float* __restrict__ omap) {
  const int b = blockIdx.y;
  const int m = blockIdx.x * 256 + threadIdx.x;
  float s = 0.f;
  for (int h = 0; h < HEADS; ++h) {
    const float* base = attn + ((size_t)(b * HEADS + h) * NN) + m;
    for (int n = 0; n < NPIX; ++n) s += base[(size_t)n * NPIX];
  }
  omap[b * NPIX + m] = s * (1.0f / (HEADS * (float)NPIX));
}

// ===========================================================================
// 6) O = P * V per (b,h): M=1024(n), K=1024(m), N=64(dd). 128x64 blocks,
//    wave w -> row-tile w, col-tiles 0..3.  grid (1, 8, 24)
// ===========================================================================
__global__ __launch_bounds__(256)
void ogemm_kernel(const float* __restrict__ attn, const float* __restrict__ Vf,
                  float* __restrict__ Oin) {
  const int bh = blockIdx.z, b = bh / HEADS, h = bh % HEADS;
  const float* P = attn + (size_t)bh * NN;                       // [n][m]
  const float* V = Vf + (size_t)(b * INNER + h * DHEAD) * NPIX;  // [dd][m]
  float* Ob = Oin + (size_t)b * NPIX * INNER + h * DHEAD;        // [n][i]
  const int n0 = blockIdx.y * 128;
  __shared__ __align__(32) bf16_t As[2][8 * 512];
  __shared__ __align__(32) bf16_t Bs[2][4 * 512];
  const int wave = threadIdx.x >> 5;
  v8f acc[4] = {};

  for (int kk = 0; kk < NPIX; kk += 64) {
    if (kk + 64 < NPIX) __builtin_prefetch(&P[(size_t)n0 * NPIX + kk + 64], 0, 1);
    stage_A<8>(As[0], P + (size_t)n0 * NPIX + kk, NPIX);
    stage_A<8>(As[1], P + (size_t)n0 * NPIX + kk + 32, NPIX);
    stage_B_kcontig<4>(Bs[0], V + kk, NPIX);
    stage_B_kcontig<4>(Bs[1], V + kk + 32, NPIX);
    __syncthreads();
#pragma unroll
    for (int kp = 0; kp < 2; ++kp)
#pragma unroll
      for (int c = 0; c < 4; ++c)
        acc[c] = wmma_frag(As[kp] + wave * 512, Bs[kp] + c * 512, acc[c]);
    __syncthreads();
  }
#pragma unroll
  for (int c = 0; c < 4; ++c)
    store_tile_rm(Ob + (size_t)(n0 + wave * 16) * INNER + c * 16, INNER, acc[c]);
}

// ===========================================================================
// 7) out_proj = Wo * Oin + bo.  Oin is [b][n][i] (k-contig B). grid (8, 3, 4)
// ===========================================================================
__global__ __launch_bounds__(256)
void outproj_kernel(const float* __restrict__ Wo, const float* __restrict__ bo,
                    const float* __restrict__ Oin, float* __restrict__ Out) {
  const int b = blockIdx.z;
  const float* X = Oin + (size_t)b * NPIX * INNER;   // [n][i]
  float* C = Out + (size_t)b * CDIM * NPIX;
  const int m0 = blockIdx.y * 128, n0 = blockIdx.x * 128;
  __shared__ __align__(32) bf16_t As[2][8 * 512];
  __shared__ __align__(32) bf16_t Bs[2][8 * 512];
  const int wave = threadIdx.x >> 5;
  const int wr0 = (wave & 3) * 2, wc0 = (wave >> 2) * 4;
  v8f acc[2][4] = {};

  for (int kk = 0; kk < INNER; kk += 64) {
    stage_A<8>(As[0], Wo + (size_t)m0 * INNER + kk, INNER);
    stage_A<8>(As[1], Wo + (size_t)m0 * INNER + kk + 32, INNER);
    stage_B_kcontig<8>(Bs[0], X + (size_t)n0 * INNER + kk, INNER);
    stage_B_kcontig<8>(Bs[1], X + (size_t)n0 * INNER + kk + 32, INNER);
    __syncthreads();
#pragma unroll
    for (int kp = 0; kp < 2; ++kp)
#pragma unroll
      for (int r = 0; r < 2; ++r)
#pragma unroll
        for (int c = 0; c < 4; ++c)
          acc[r][c] = wmma_frag(As[kp] + (wr0 + r) * 512,
                                Bs[kp] + (wc0 + c) * 512, acc[r][c]);
    __syncthreads();
  }
  const int lane = threadIdx.x & 31, col = lane & 15, r0 = (lane >> 4) * 8;
#pragma unroll
  for (int r = 0; r < 2; ++r)
#pragma unroll
    for (int c = 0; c < 4; ++c) {
      const int mb = m0 + (wr0 + r) * 16, nb = n0 + (wc0 + c) * 16;
#pragma unroll
      for (int i = 0; i < 8; ++i)
        C[(size_t)(mb + r0 + i) * NPIX + nb + col] = acc[r][c][i] + bo[mb + r0 + i];
    }
}

// ===========================================================================
extern "C" void kernel_launch(void* const* d_in, const int* in_sizes, int n_in,
                              void* d_out, int out_size, void* d_ws, size_t ws_size,
                              hipStream_t stream) {
  (void)in_sizes; (void)n_in; (void)out_size; (void)ws_size;
  const float* query   = (const float*)d_in[0];
  const float* context = (const float*)d_in[1];
  const float* Wq = (const float*)d_in[2];
  const float* Wk = (const float*)d_in[3];
  const float* Wv = (const float*)d_in[4];
  const float* Wo = (const float*)d_in[5];
  const float* bo = (const float*)d_in[6];

  // ---- d_out layout (flat, reference return order) ----
  float* out = (float*)d_out;
  const size_t SZ_PROJ = (size_t)BATCH * CDIM * NPIX;      // 1,572,864
  const size_t SZ_MAP  = (size_t)BATCH * NPIX;             // 4,096
  const size_t SZ_ATTN = (size_t)BH * NN;                  // 25,165,824
  float* out_proj      = out;
  float* out_map       = out + SZ_PROJ;
  float* out_causal    = out + SZ_PROJ + SZ_MAP;
  float* out_noncausal = out_causal + SZ_ATTN;
  float* out_bmask     = out_noncausal + SZ_ATTN;

  // ---- workspace layout (needs ~126 MB) ----
  float* Qf   = (float*)d_ws;            // [bh][token][dd]
  float* Kf   = Qf + SZ_PROJ;            // [bh][token][dd]
  float* Vf   = Kf + SZ_PROJ;            // [b][i][m]
  float* attn = Vf + SZ_PROJ;            // [bh][n][m]
  float* Oin  = attn + SZ_ATTN;          // [b][n][i]
  unsigned* hist = (unsigned*)(Oin + SZ_PROJ);
  unsigned* pref = hist + BH * 256;
  unsigned* want = pref + BH;
  unsigned* tie  = want + BH;

  const dim3 blk(256);
  proj_gemm_kernel<<<dim3(8, 3, 12), blk, 0, stream>>>(Wq, Wk, Wv, query, context,
                                                       Qf, Kf, Vf);
  scores_kernel<<<dim3(8, 8, BH), blk, 0, stream>>>(Qf, Kf, attn);
  softmax_kernel<<<dim3(BH * NPIX), blk, 0, stream>>>(attn);

  topk_init_kernel<<<dim3(BH), blk, 0, stream>>>(hist, pref, want, tie);
  for (int p = 0; p < 4; ++p) {
    const int shift = 24 - 8 * p;
    const unsigned maskTop = (p == 0) ? 0u : (0xFFFFFFFFu << (unsigned)(shift + 8));
    topk_hist_kernel<<<dim3(64, BH), blk, 0, stream>>>(attn, hist, pref, maskTop, shift);
    topk_scan_kernel<<<dim3(BH), blk, 0, stream>>>(hist, pref, want, shift);
  }
  topk_mask_kernel<<<dim3(64, BH), blk, 0, stream>>>(attn, pref, want, tie,
                                                     out_causal, out_noncausal,
                                                     out_bmask);
  attnmap_kernel<<<dim3(4, BATCH), blk, 0, stream>>>(attn, out_map);
  ogemm_kernel<<<dim3(1, 8, BH), blk, 0, stream>>>(attn, Vf, Oin);
  outproj_kernel<<<dim3(8, 3, BATCH), blk, 0, stream>>>(Wo, bo, Oin, out_proj);
}
